// AttentionBlock_37761352466542
// MI455X (gfx1250) — matrix operand order repaired and checked by reading, not compiled, not run on hardware
//
#include <hip/hip_runtime.h>

// ---------------------------------------------------------------------------
// Types for CDNA5 WMMA (wave32, 16x16x32 bf16 -> f32)
// ---------------------------------------------------------------------------
typedef __attribute__((ext_vector_type(16))) __bf16 v16bf;
typedef __attribute__((ext_vector_type(8)))  float  v8f;

union FragBF16 {
    uint4 q[2];   // 32 bytes = 16 bf16
    v16bf v;
};

__device__ __forceinline__ unsigned short f2bf(float f) {
    union { float f; unsigned int u; } x; x.f = f;
    unsigned int u = x.u;
    u += 0x7FFFu + ((u >> 16) & 1u);   // round-to-nearest-even
    return (unsigned short)(u >> 16);
}
__device__ __forceinline__ float bf2f(unsigned short h) {
    union { unsigned int u; float f; } x; x.u = ((unsigned int)h) << 16;
    return x.f;
}

// ---------------------------------------------------------------------------
// Weight repack: (Cout, Cin, kh, kw) f32  ->  [tap][co][ci] bf16
// ---------------------------------------------------------------------------
__global__ void repack_w_kernel(const float* __restrict__ w,
                                unsigned short* __restrict__ wp,
                                int Cout, int Cin, int taps) {
    size_t i = (size_t)blockIdx.x * blockDim.x + threadIdx.x;
    size_t tot = (size_t)Cout * Cin * taps;
    if (i >= tot) return;
    int ci = (int)(i % Cin);
    size_t r = i / Cin;
    int co = (int)(r % Cout);
    int t  = (int)(r / Cout);
    wp[i] = f2bf(w[((size_t)co * Cin + ci) * taps + t]);
}

// ---------------------------------------------------------------------------
// Layer-0 direct conv (Cin=3).  out: [co][BHW] f32
// ---------------------------------------------------------------------------
__global__ void conv0_kernel(const float* __restrict__ x,
                             const float* __restrict__ w,
                             float* __restrict__ out,
                             int Bn, int H, int W) {
    size_t BHW = (size_t)Bn * H * W;
    size_t i = (size_t)blockIdx.x * blockDim.x + threadIdx.x;
    if (i >= 64 * BHW) return;
    int co = (int)(i / BHW);
    size_t p = i % BHW;
    int b  = (int)(p / ((size_t)H * W));
    int r  = (int)(p % ((size_t)H * W));
    int y  = r / W, xq = r % W;
    float s = 0.f;
    #pragma unroll
    for (int ci = 0; ci < 3; ++ci)
        #pragma unroll
        for (int ky = 0; ky < 3; ++ky) {
            int yy = y + ky - 1;
            if (yy < 0 || yy >= H) continue;
            #pragma unroll
            for (int kx = 0; kx < 3; ++kx) {
                int xx = xq + kx - 1;
                if (xx < 0 || xx >= W) continue;
                s += w[((co * 3 + ci) * 3 + ky) * 3 + kx] *
                     x[(((size_t)b * 3 + ci) * H + yy) * W + xx];
            }
        }
    out[(size_t)co * BHW + p] = s;
}

// ---------------------------------------------------------------------------
// Implicit-GEMM conv / GEMM via WMMA bf16 with 2x4 register blocking.
//   act : NHWC bf16  [B*H*W][Cin]
//   wp  : [taps][Cout][Cin] bf16
//   out_cp   : f32 [Cout][BHW]        (conv path, bias == nullptr)
//   out_nhwc : f32 [BHW][Cout] + bias (QKV path)
// One wave = 32(M=Cout) x 64(N=spatial) output block = 8 accumulators.
// Per 32-K chunk: 12 b128 loads feed 8 WMMAs (1.5 loads/wmma).
// Fragment layouts per CDNA5 ISA 7.12.2 (16-bit A 16x32, B 32x16).
// ---------------------------------------------------------------------------
__global__ __launch_bounds__(256)
void wmma_conv_kernel(const unsigned short* __restrict__ act,
                      const unsigned short* __restrict__ wp,
                      const float* __restrict__ bias,
                      float* __restrict__ out_cp,
                      float* __restrict__ out_nhwc,
                      int Cin, int Cout, int Bn, int H, int W, int taps) {
    const int lane = threadIdx.x & 31;
    const int wave = threadIdx.x >> 5;
    const int HW   = H * W;
    const int BHW  = Bn * HW;
    const int nbn  = BHW >> 6;           // 64-wide N blocks
    const int nbm  = Cout >> 5;          // 32-tall M blocks
    int blkid = blockIdx.x * (blockDim.x >> 5) + wave;
    if (blkid >= nbm * nbn) return;
    int mb = blkid / nbn, nb = blkid % nbn;
    int co0 = mb << 5, p0 = nb << 6;

    const int hl = lane >> 4;            // lane half
    const int lm = lane & 15;
    const int kbA = hl * 8;              // A k-base within chunk
    const int kbB = hl * 16;             // unused (B loads also use kbA)

    // this lane's four spatial columns
    int py[4], px[4], pb[4];
    #pragma unroll
    for (int j = 0; j < 4; ++j) {
        int p = p0 + j * 16 + lm;
        pb[j] = p / HW;
        int rr = p % HW;
        py[j] = rr / W;
        px[j] = rr % W;
    }

    v8f acc[2][4] = {};
    const unsigned short* arow0 = wp + (size_t)(co0 + lm) * Cin;
    const unsigned short* arow1 = wp + (size_t)(co0 + 16 + lm) * Cin;
    const size_t tapStride = (size_t)Cout * Cin;

    for (int t = 0; t < taps; ++t) {
        int dy = (taps == 9) ? (t / 3 - 1) : 0;
        int dx = (taps == 9) ? (t % 3 - 1) : 0;
        const unsigned short* aptr0 = arow0 + (size_t)t * tapStride;
        const unsigned short* aptr1 = arow1 + (size_t)t * tapStride;
        const unsigned short* bptr[4];
        bool inb[4];
        #pragma unroll
        for (int j = 0; j < 4; ++j) {
            int yy = py[j] + dy, xx = px[j] + dx;
            inb[j] = (yy >= 0) && (yy < H) && (xx >= 0) && (xx < W);
            bptr[j] = inb[j]
                ? act + ((size_t)pb[j] * HW + (size_t)yy * W + xx) * Cin
                : (const unsigned short*)0;
        }

        for (int c0 = 0; c0 < Cin; c0 += 32) {
            FragBF16 fa0, fa1, fb[4];
            fa0.q[0] = *(const uint4*)(aptr0 + c0 + kbA);
            fa0.q[1] = *(const uint4*)(aptr0 + c0 + 16 + kbA);
            fa1.q[0] = *(const uint4*)(aptr1 + c0 + kbA);
            fa1.q[1] = *(const uint4*)(aptr1 + c0 + 16 + kbA);
            #pragma unroll
            for (int j = 0; j < 4; ++j) {
                if (inb[j]) {
                    fb[j].q[0] = *(const uint4*)(bptr[j] + c0 + kbA);
                    fb[j].q[1] = *(const uint4*)(bptr[j] + c0 + 16 + kbA);
                } else {
                    fb[j].q[0] = make_uint4(0u, 0u, 0u, 0u);
                    fb[j].q[1] = make_uint4(0u, 0u, 0u, 0u);
                }
            }
            #pragma unroll
            for (int j = 0; j < 4; ++j) {
                acc[0][j] = __builtin_amdgcn_wmma_f32_16x16x32_bf16(
                    false, fa0.v, false, fb[j].v, (short)0, acc[0][j], false, false);
                acc[1][j] = __builtin_amdgcn_wmma_f32_16x16x32_bf16(
                    false, fa1.v, false, fb[j].v, (short)0, acc[1][j], false, false);
            }
        }
    }

    // C/D layout: VGPR r, lanes0-15 -> M=r, lanes16-31 -> M=8+r; N = lane&15
    if (out_nhwc) {
        #pragma unroll
        for (int i = 0; i < 2; ++i)
            #pragma unroll
            for (int j = 0; j < 4; ++j)
                #pragma unroll
                for (int r = 0; r < 8; ++r) {
                    int co = co0 + i * 16 + kbA + r;
                    int p  = p0 + j * 16 + lm;
                    out_nhwc[(size_t)p * Cout + co] =
                        acc[i][j][r] + (bias ? bias[co] : 0.f);
                }
    } else {
        #pragma unroll
        for (int i = 0; i < 2; ++i)
            #pragma unroll
            for (int j = 0; j < 4; ++j)
                #pragma unroll
                for (int r = 0; r < 8; ++r) {
                    int co = co0 + i * 16 + kbA + r;
                    int p  = p0 + j * 16 + lm;
                    out_cp[(size_t)co * BHW + p] = acc[i][j][r];
                }
    }
}

// ---------------------------------------------------------------------------
// BatchNorm statistics: one block per channel over [co][BHW] f32
// ---------------------------------------------------------------------------
__global__ void bn_stats_kernel(const float* __restrict__ o,
                                float* __restrict__ stats, int BHW) {
    int c = blockIdx.x;
    const float* p = o + (size_t)c * BHW;
    float s = 0.f, ss = 0.f;
    for (int i = threadIdx.x; i < BHW; i += blockDim.x) {
        float v = p[i]; s += v; ss += v * v;
    }
    __shared__ float sh0[256], sh1[256];
    sh0[threadIdx.x] = s; sh1[threadIdx.x] = ss;
    __syncthreads();
    for (int st = 128; st > 0; st >>= 1) {
        if ((int)threadIdx.x < st) {
            sh0[threadIdx.x] += sh0[threadIdx.x + st];
            sh1[threadIdx.x] += sh1[threadIdx.x + st];
        }
        __syncthreads();
    }
    if (threadIdx.x == 0) {
        float inv = 1.f / (float)BHW;
        float m = sh0[0] * inv;
        float var = sh1[0] * inv - m * m;
        stats[2 * c]     = m;
        stats[2 * c + 1] = rsqrtf(var + 1e-5f);
    }
}

// ---------------------------------------------------------------------------
// BN apply + ReLU: read [co][BHW] f32, write NHWC bf16 [p][C]
// ---------------------------------------------------------------------------
__global__ void bn_apply_kernel(const float* __restrict__ o,
                                const float* __restrict__ stats,
                                const float* __restrict__ g,
                                const float* __restrict__ bb,
                                unsigned short* __restrict__ act,
                                int C, int BHW) {
    size_t i = (size_t)blockIdx.x * blockDim.x + threadIdx.x;
    if (i >= (size_t)C * BHW) return;
    int c = (int)(i % C);
    size_t p = i / C;
    float m = stats[2 * c], rs = stats[2 * c + 1];
    float v = (o[(size_t)c * BHW + p] - m) * rs * g[c] + bb[c];
    v = fmaxf(v, 0.f);
    act[(size_t)p * C + c] = f2bf(v);
}

// ---------------------------------------------------------------------------
// 2x2 max pool on NHWC bf16
// ---------------------------------------------------------------------------
__global__ void pool_kernel(const unsigned short* __restrict__ in,
                            unsigned short* __restrict__ out,
                            int C, int Bn, int Hi, int Wi) {
    int Ho = Hi >> 1, Wo = Wi >> 1;
    size_t tot = (size_t)Bn * Ho * Wo * C;
    size_t i = (size_t)blockIdx.x * blockDim.x + threadIdx.x;
    if (i >= tot) return;
    int c = (int)(i % C);
    size_t r = i / C;
    int xo = (int)(r % Wo); r /= Wo;
    int yo = (int)(r % Ho);
    int b  = (int)(r / Ho);
    size_t rowb = ((size_t)b * Hi + 2 * yo) * Wi;
    float v00 = bf2f(in[(rowb + 2 * xo) * C + c]);
    float v01 = bf2f(in[(rowb + 2 * xo + 1) * C + c]);
    float v10 = bf2f(in[(rowb + Wi + 2 * xo) * C + c]);
    float v11 = bf2f(in[(rowb + Wi + 2 * xo + 1) * C + c]);
    out[i] = f2bf(fmaxf(fmaxf(v00, v01), fmaxf(v10, v11)));
}

// ---------------------------------------------------------------------------
// rel[ky*7+kx] = sum_c rel_h[c,ky,0] + sum_c rel_w[c,0,kx]   (c = 0..1023)
// ---------------------------------------------------------------------------
__global__ void rel_kernel(const float* __restrict__ rel_h,
                           const float* __restrict__ rel_w,
                           float* __restrict__ rel) {
    int t = threadIdx.x;
    if (t >= 49) return;
    int ky = t / 7, kx = t % 7;
    float sh = 0.f, sw = 0.f;
    for (int c = 0; c < 1024; ++c) {
        sh += rel_h[c * 7 + ky];
        sw += rel_w[c * 7 + kx];
    }
    rel[t] = sh + sw;
}

// ---------------------------------------------------------------------------
// Attention with the reference's scrambled reshapes made explicit.
// Group t (0..255) = channels [8t, 8t+8) x all 64 spatial positions.
// Block = (b, t). attn flat per batch index m = t*512 + e,
//   e -> co = 8t + (e>>6), y = (e>>3)&7, x = e&7.
// OOB patch positions use k = kb (bias), v = vb.
// ---------------------------------------------------------------------------
__global__ __launch_bounds__(256)
void attn_kernel(const float* __restrict__ Q,
                 const float* __restrict__ Kk,
                 const float* __restrict__ V,
                 const float* __restrict__ kb,
                 const float* __restrict__ vb,
                 const float* __restrict__ rel,
                 float* __restrict__ attn) {
    const int C = 2048;
    int blk = blockIdx.x;
    int t = blk & 255;
    int b = blk >> 8;
    int G = t << 3;

    __shared__ float part[256 * 49];
    __shared__ float red[256];
    __shared__ float s_qk[49];
    __shared__ float s_qsum;

    float pj[49];
    #pragma unroll
    for (int j = 0; j < 49; ++j) pj[j] = 0.f;
    float qs = 0.f;

    for (int e = threadIdx.x; e < 512; e += 256) {
        int co = G + (e >> 6);
        int y = (e >> 3) & 7, x = e & 7;
        float qv = Q[(size_t)((b * 8 + y) * 8 + x) * C + co];
        qs += qv;
        #pragma unroll
        for (int j = 0; j < 49; ++j) {
            int dy = j / 7 - 3, dx = j % 7 - 3;
            int yy = y + dy, xx = x + dx;
            float kv = (yy >= 0 && yy < 8 && xx >= 0 && xx < 8)
                           ? Kk[(size_t)((b * 8 + yy) * 8 + xx) * C + co]
                           : kb[co];
            pj[j] += qv * kv;
        }
    }
    #pragma unroll
    for (int j = 0; j < 49; ++j) part[threadIdx.x * 49 + j] = pj[j];
    red[threadIdx.x] = qs;
    __syncthreads();

    if (threadIdx.x < 49) {
        float s = 0.f;
        for (int i = 0; i < 256; ++i) s += part[i * 49 + threadIdx.x];
        s_qk[threadIdx.x] = s;
    }
    if (threadIdx.x == 0) {
        float q0 = 0.f;
        for (int i = 0; i < 256; ++i) q0 += red[i];
        s_qsum = q0;
    }
    __syncthreads();

    if (threadIdx.x == 0) {
        float mx = -1e30f;
        #pragma unroll
        for (int j = 0; j < 49; ++j) {
            s_qk[j] += s_qsum * rel[j];
            mx = fmaxf(mx, s_qk[j]);
        }
        float sm = 0.f;
        #pragma unroll
        for (int j = 0; j < 49; ++j) {
            float ex = __expf(s_qk[j] - mx);
            s_qk[j] = ex; sm += ex;
        }
        float inv = 1.f / sm;
        #pragma unroll
        for (int j = 0; j < 49; ++j) s_qk[j] *= inv;
    }
    __syncthreads();

    for (int e = threadIdx.x; e < 512; e += 256) {
        int co = G + (e >> 6);
        int y = (e >> 3) & 7, x = e & 7;
        float o = 0.f;
        #pragma unroll
        for (int j = 0; j < 49; ++j) {
            int dy = j / 7 - 3, dx = j % 7 - 3;
            int yy = y + dy, xx = x + dx;
            float vv = (yy >= 0 && yy < 8 && xx >= 0 && xx < 8)
                           ? V[(size_t)((b * 8 + yy) * 8 + xx) * C + co]
                           : vb[co];
            o += s_qk[j] * vv;
        }
        attn[(size_t)b * 131072 + (size_t)t * 512 + e] = o;
    }
}

// ---------------------------------------------------------------------------
// Final channel softmax: for fixed (b, pos=y*8+x), softmax over the 2048
// values at attn[b*131072 + c*64 + pos]; output in the same (NCHW) layout.
// ---------------------------------------------------------------------------
__global__ void softmax_ch_kernel(const float* __restrict__ attn,
                                  float* __restrict__ out) {
    int s = blockIdx.x;
    int b = s >> 6, pos = s & 63;
    const float* a = attn + (size_t)b * 131072 + pos;
    float* o = out + (size_t)b * 131072 + pos;
    __shared__ float red[256];

    float mx = -1e30f;
    for (int c = threadIdx.x; c < 2048; c += 256)
        mx = fmaxf(mx, a[(size_t)c * 64]);
    red[threadIdx.x] = mx;
    __syncthreads();
    for (int st = 128; st > 0; st >>= 1) {
        if ((int)threadIdx.x < st)
            red[threadIdx.x] = fmaxf(red[threadIdx.x], red[threadIdx.x + st]);
        __syncthreads();
    }
    mx = red[0];
    __syncthreads();

    float sm = 0.f;
    for (int c = threadIdx.x; c < 2048; c += 256)
        sm += __expf(a[(size_t)c * 64] - mx);
    red[threadIdx.x] = sm;
    __syncthreads();
    for (int st = 128; st > 0; st >>= 1) {
        if ((int)threadIdx.x < st)
            red[threadIdx.x] += red[threadIdx.x + st];
        __syncthreads();
    }
    float inv = 1.f / red[0];
    for (int c = threadIdx.x; c < 2048; c += 256)
        o[(size_t)c * 64] = __expf(a[(size_t)c * 64] - mx) * inv;
}

// ---------------------------------------------------------------------------
// Host orchestration
// ---------------------------------------------------------------------------
extern "C" void kernel_launch(void* const* d_in, const int* in_sizes, int n_in,
                              void* d_out, int out_size, void* d_ws, size_t ws_size,
                              hipStream_t stream) {
    (void)in_sizes; (void)n_in; (void)out_size; (void)ws_size;
    const float* input_x = (const float*)d_in[0];
    const float *wN[12], *gN[12], *bN[12];
    for (int n = 0; n < 12; ++n) {
        wN[n] = (const float*)d_in[1 + 3 * n];
        gN[n] = (const float*)d_in[2 + 3 * n];
        bN[n] = (const float*)d_in[3 + 3 * n];
    }
    const float* kw    = (const float*)d_in[37];
    const float* kbias = (const float*)d_in[38];
    const float* qw    = (const float*)d_in[39];
    const float* qbias = (const float*)d_in[40];
    const float* vw    = (const float*)d_in[41];
    const float* vbias = (const float*)d_in[42];
    const float* rel_h = (const float*)d_in[43];
    const float* rel_w = (const float*)d_in[44];

    char* base = (char*)d_ws;
    size_t off = 0;
    auto alloc = [&](size_t bytes) -> void* {
        void* p = base + off;
        off = (off + bytes + 255) & ~(size_t)255;
        return p;
    };
    unsigned short* WP = (unsigned short*)alloc((size_t)2048 * 2048 * 9 * 2); // 75.5 MB
    float* OT          = (float*)alloc((size_t)64 * 262144 * 4);              // 64 MB
    unsigned short* A1 = (unsigned short*)alloc((size_t)64 * 262144 * 2);     // 32 MB
    unsigned short* A2 = (unsigned short*)alloc((size_t)64 * 262144 * 2);     // 32 MB
    float* STATS       = (float*)alloc((size_t)2048 * 2 * 4);
    float* Qb          = (float*)alloc((size_t)256 * 2048 * 4);
    float* Kb          = (float*)alloc((size_t)256 * 2048 * 4);
    float* Vb          = (float*)alloc((size_t)256 * 2048 * 4);
    float* ATTN        = (float*)alloc((size_t)4 * 131072 * 4);
    float* REL         = (float*)alloc(64 * 4);

    const int Bn = 4;
    int H = 256, W = 256;

    // ---- layer 0 (Cin=3, direct) ----
    {
        int BHW = Bn * H * W;
        size_t tot = (size_t)64 * BHW;
        conv0_kernel<<<(unsigned)((tot + 255) / 256), 256, 0, stream>>>(input_x, wN[0], OT, Bn, H, W);
        bn_stats_kernel<<<64, 256, 0, stream>>>(OT, STATS, BHW);
        bn_apply_kernel<<<(unsigned)((tot + 255) / 256), 256, 0, stream>>>(OT, STATS, gN[0], bN[0], A1, 64, BHW);
    }
    unsigned short* cur = A1;
    unsigned short* nxt = A2;

    auto conv_bn = [&](const unsigned short* in_, unsigned short* out_,
                       int n, int Cin, int Cout, int Hc, int Wc) {
        int BHW = Bn * Hc * Wc;
        size_t wtot = (size_t)Cout * Cin * 9;
        repack_w_kernel<<<(unsigned)((wtot + 255) / 256), 256, 0, stream>>>(wN[n], WP, Cout, Cin, 9);
        int tiles = (Cout / 32) * (BHW / 64);
        wmma_conv_kernel<<<(unsigned)((tiles + 7) / 8), 256, 0, stream>>>(
            in_, WP, (const float*)0, OT, (float*)0, Cin, Cout, Bn, Hc, Wc, 9);
        bn_stats_kernel<<<Cout, 256, 0, stream>>>(OT, STATS, BHW);
        size_t tot = (size_t)Cout * BHW;
        bn_apply_kernel<<<(unsigned)((tot + 255) / 256), 256, 0, stream>>>(OT, STATS, gN[n], bN[n], out_, Cout, BHW);
    };

    // ---- layer 1 @256 ----
    conv_bn(cur, nxt, 1, 64, 64, H, W);
    { unsigned short* t = cur; cur = nxt; nxt = t; }

    const int chans[6] = {64, 128, 256, 512, 1024, 2048};
    int Cprev = 64;
    for (int s = 1; s < 6; ++s) {
        // pool
        {
            int Ho = H / 2, Wo = W / 2;
            size_t tot = (size_t)Bn * Ho * Wo * Cprev;
            pool_kernel<<<(unsigned)((tot + 255) / 256), 256, 0, stream>>>(cur, nxt, Cprev, Bn, H, W);
            { unsigned short* t = cur; cur = nxt; nxt = t; }
            H = Ho; W = Wo;
        }
        int n = 2 * s;
        int Cout = chans[s];
        conv_bn(cur, nxt, n, Cprev, Cout, H, W);
        { unsigned short* t = cur; cur = nxt; nxt = t; }
        conv_bn(cur, nxt, n + 1, Cout, Cout, H, W);
        { unsigned short* t = cur; cur = nxt; nxt = t; }
        Cprev = Cout;
    }

    // ---- QKV 1x1 GEMMs (H=W=8, C=2048) ----
    {
        int BHW = Bn * 64;
        int tiles = (2048 / 32) * (BHW / 64);
        size_t wtot = (size_t)2048 * 2048;
        unsigned wblk = (unsigned)((wtot + 255) / 256);
        unsigned gblk = (unsigned)((tiles + 7) / 8);

        repack_w_kernel<<<wblk, 256, 0, stream>>>(qw, WP, 2048, 2048, 1);
        wmma_conv_kernel<<<gblk, 256, 0, stream>>>(cur, WP, qbias, (float*)0, Qb, 2048, 2048, Bn, 8, 8, 1);

        repack_w_kernel<<<wblk, 256, 0, stream>>>(kw, WP, 2048, 2048, 1);
        wmma_conv_kernel<<<gblk, 256, 0, stream>>>(cur, WP, kbias, (float*)0, Kb, 2048, 2048, Bn, 8, 8, 1);

        repack_w_kernel<<<wblk, 256, 0, stream>>>(vw, WP, 2048, 2048, 1);
        wmma_conv_kernel<<<gblk, 256, 0, stream>>>(cur, WP, vbias, (float*)0, Vb, 2048, 2048, Bn, 8, 8, 1);
    }

    // ---- attention + final softmax ----
    rel_kernel<<<1, 64, 0, stream>>>(rel_h, rel_w, REL);
    attn_kernel<<<1024, 256, 0, stream>>>(Qb, Kb, Vb, kbias, vbias, REL, ATTN);
    softmax_ch_kernel<<<256, 256, 0, stream>>>(ATTN, (float*)d_out);
}
